// AttentionDTI_31482110279973
// MI455X (gfx1250) — matrix operand order repaired and checked by reading, not compiled, not run on hardware
//
#include <hip/hip_runtime.h>
#include <hip/hip_bf16.h>

typedef __attribute__((ext_vector_type(16))) _Float16 v16h;
typedef __attribute__((ext_vector_type(8)))  _Float16 v8h;
typedef __attribute__((ext_vector_type(8)))  float    v8f;

// ---------------- problem constants ----------------
#define BB    8
#define LD    100
#define LP    1000
#define DIMD  64
#define CONVC 40
#define C4    160
#define DL1   97
#define DL2   92
#define DL3   85
#define PL1   997
#define PL2   990
#define PL3   979

#define ACT_NONE 0
#define ACT_RELU 1
#define ACT_SIGM 2
#define ACT_LREL 3

__device__ __forceinline__ float activate(float v, int act) {
    switch (act) {
        case ACT_RELU: return v > 0.f ? v : 0.f;
        case ACT_SIGM: return 1.f / (1.f + __expf(-v));
        case ACT_LREL: return v > 0.f ? v : 0.01f * v;
        default:       return v;
    }
}

// ---------------- elementwise helpers ----------------

__global__ void embed_gather(const int* __restrict__ idx,
                             const float* __restrict__ emb,
                             _Float16* __restrict__ out,
                             int total /*B*L*/, int dim) {
    int t = blockIdx.x * blockDim.x + threadIdx.x;
    if (t >= total * dim) return;
    int pos = t / dim, c = t - pos * dim;
    out[t] = (_Float16)emb[(size_t)idx[pos] * dim + c];
}

// ---------------- B-matrix packers: fragment-major layout ----------------
// Packed layout: Bp[kt][tn][lane][e], e=0..15 contiguous per lane (32B).
// Element (lane,e) of tile (kt,tn): n = tn*16 + (lane&15), kk = kt*32 + (e&7) + 8*(lane>>4) + 16*(e>>3).
// K/N pad positions are stored as 0 so unguarded A loads are harmless.

// Dense W [K,N] row-major f32 -> packed f16
__global__ void pack_w_dense(const float* __restrict__ W,
                             _Float16* __restrict__ Bp,
                             int K, int N, int KT, int NT) {
    int t = blockIdx.x * blockDim.x + threadIdx.x;
    int total = KT * NT * 512;
    if (t >= total) return;
    int kt   = t / (NT * 512);
    int rem  = t - kt * (NT * 512);
    int tn   = rem / 512;
    int r2   = rem - tn * 512;
    int lane = r2 >> 4;
    int e    = r2 & 15;
    int n  = tn * 16 + (lane & 15);
    int kk = kt * 32 + (e & 7) + 8 * (lane >> 4) + 16 * (e >> 3);
    Bp[t] = (kk < K && n < N) ? (_Float16)W[(size_t)kk * N + n] : (_Float16)0.f;
}

// Conv weights w[Cout][Cin][Kw] (OIH) -> packed over kk=k*Cin+ci, N=Cout
__global__ void pack_w_conv(const float* __restrict__ w,
                            _Float16* __restrict__ Bp,
                            int Cout, int Cin, int Kw, int KT, int NT) {
    int t = blockIdx.x * blockDim.x + threadIdx.x;
    int total = KT * NT * 512;
    if (t >= total) return;
    int Kd   = Kw * Cin;
    int kt   = t / (NT * 512);
    int rem  = t - kt * (NT * 512);
    int tn   = rem / 512;
    int r2   = rem - tn * 512;
    int lane = r2 >> 4;
    int e    = r2 & 15;
    int co = tn * 16 + (lane & 15);
    int kk = kt * 32 + (e & 7) + 8 * (lane >> 4) + 16 * (e >> 3);
    _Float16 v = (_Float16)0.f;
    if (kk < Kd && co < Cout) {
        int k  = kk / Cin;
        int ci = kk - k * Cin;
        v = (_Float16)w[((size_t)co * Cin + ci) * Kw + k];
    }
    Bp[t] = v;
}

// ---------------- WMMA GEMM core (templated compile-time K) ----------------
// A [M,KD] row-major f16 (rows beyond M readable garbage; never stored).
// Bp packed fragment-major. One wave per 16x16 C tile, 8 waves/block.
template <int KD>
__global__ void gemm_f16_wmma(const _Float16* __restrict__ A,
                              const _Float16* __restrict__ Bp,
                              const float* __restrict__ bias,
                              float* __restrict__ out32,
                              _Float16* __restrict__ out16,
                              int M, int N, int act) {
    constexpr int KT = (KD + 31) / 32;
    int wave = threadIdx.x >> 5;
    int lane = threadIdx.x & 31;
    int tilesM = (M + 15) >> 4;
    int tilesN = (N + 15) >> 4;
    int tile = blockIdx.x * 8 + wave;
    if (tile >= tilesM * tilesN) return;      // uniform per wave
    int tm = tile % tilesM;
    int tn = tile / tilesM;
    int half = lane >> 4, l15 = lane & 15;

    int mA = tm * 16 + l15;
    const _Float16* ap = A + (size_t)mA * KD + 8 * half;
    const _Float16* bp = Bp + ((size_t)tn * 32 + lane) * 16;
    const size_t bstep = (size_t)tilesN * 512;

    v8f c = {};
    for (int kt = 0; kt < KT; ++kt) {
        v8h lo = *(const v8h*)(ap);           // K = k0+8h .. +7   (b128)
        v8h hi = *(const v8h*)(ap + 16);      // K = k0+16+8h .. +7 (b128)
        v16h b = *(const v16h*)(bp);          // whole B fragment   (2x b128)
        v16h a = __builtin_shufflevector(lo, hi, 0,1,2,3,4,5,6,7,8,9,10,11,12,13,14,15);
        c = __builtin_amdgcn_wmma_f32_16x16x32_f16(false, a, false, b,
                                                   (short)0, c, false, false);
        ap += 32;
        bp += bstep;
    }
#pragma unroll
    for (int r = 0; r < 8; ++r) {
        int m = tm * 16 + r + 8 * half;
        int n = tn * 16 + l15;
        if (m < M && n < N) {
            float v = activate(c[r] + bias[n], act);
            if (out32) out32[(size_t)m * N + n] = v;
            if (out16) out16[(size_t)m * N + n] = (_Float16)v;
        }
    }
}

// ---------------- implicit-GEMM conv1d (channel-last) + ReLU ----------------
// In [B, Lin, Cin] f16; im2col row (b,l) = contiguous slice of Kw*Cin halves.
template <int KD>
__global__ void conv_gemm_wmma(const _Float16* __restrict__ In,
                               const _Float16* __restrict__ Bp,
                               const float* __restrict__ bias,
                               _Float16* __restrict__ out16,
                               float* __restrict__ out32,
                               int Bn, int Lin, int Lout, int Cin, int Cout) {
    constexpr int KT = (KD + 31) / 32;
    int Mtot = Bn * Lout;
    int wave = threadIdx.x >> 5;
    int lane = threadIdx.x & 31;
    int tilesM = (Mtot + 15) >> 4;
    int tilesN = (Cout + 15) >> 4;
    int tile = blockIdx.x * 8 + wave;
    if (tile >= tilesM * tilesN) return;
    int tm = tile % tilesM;
    int tn = tile / tilesM;
    int half = lane >> 4, l15 = lane & 15;

    int mA = tm * 16 + l15;
    int bb = mA / Lout;
    int ll = mA - bb * Lout;
    const _Float16* ap = In + ((size_t)bb * Lin + ll) * Cin + 8 * half;
    const _Float16* bp = Bp + ((size_t)tn * 32 + lane) * 16;
    const size_t bstep = (size_t)tilesN * 512;

    v8f c = {};
    for (int kt = 0; kt < KT; ++kt) {
        v8h lo = *(const v8h*)(ap);
        v8h hi = *(const v8h*)(ap + 16);
        v16h b = *(const v16h*)(bp);
        v16h a = __builtin_shufflevector(lo, hi, 0,1,2,3,4,5,6,7,8,9,10,11,12,13,14,15);
        c = __builtin_amdgcn_wmma_f32_16x16x32_f16(false, a, false, b,
                                                   (short)0, c, false, false);
        ap += 32;
        bp += bstep;
    }
#pragma unroll
    for (int r = 0; r < 8; ++r) {
        int m = tm * 16 + r + 8 * half;
        int n = tn * 16 + l15;
        if (m < Mtot && n < Cout) {
            float v = c[r] + bias[n];
            v = v > 0.f ? v : 0.f;
            out16[(size_t)m * Cout + n] = (_Float16)v;
            if (out32) out32[(size_t)m * Cout + n] = v;
        }
    }
}

// ---------------- fused attention means (R never materialized) ----------------
__global__ void attn_rowmean(const float* __restrict__ datt,
                             const float* __restrict__ patt,
                             _Float16* __restrict__ rowm) {
    int bi = blockIdx.x;
    int c  = threadIdx.x;
    int b  = bi / DL3;
    float d = datt[(size_t)bi * C4 + c];
    const float* pb = patt + (size_t)b * PL3 * C4 + c;
    float s = 0.f;
    for (int j = 0; j < PL3; ++j)
        s += fmaxf(d + pb[(size_t)j * C4], 0.f);
    rowm[(size_t)bi * C4 + c] = (_Float16)(s * (1.f / (float)PL3));
}

__global__ void attn_colmean(const float* __restrict__ datt,
                             const float* __restrict__ patt,
                             _Float16* __restrict__ colm) {
    int bj = blockIdx.x;
    int c  = threadIdx.x;
    int b  = bj / PL3;
    float p = patt[(size_t)bj * C4 + c];
    const float* db = datt + (size_t)b * DL3 * C4 + c;
    float s = 0.f;
    for (int i = 0; i < DL3; ++i)
        s += fmaxf(db[(size_t)i * C4] + p, 0.f);
    colm[(size_t)bj * C4 + c] = (_Float16)(s * (1.f / (float)DL3));
}

// ---------------- gate (x*(0.5+sig)) + global max pool -> pair[B,320] ----------------
__global__ void gate_pool(const float* __restrict__ dc32,
                          const float* __restrict__ comp,
                          const float* __restrict__ pc32,
                          const float* __restrict__ prot,
                          _Float16* __restrict__ pair) {
    int b = blockIdx.x >> 1;
    int which = blockIdx.x & 1;
    int c = threadIdx.x;
    float m = -1e30f;
    if (which == 0) {
        for (int l = 0; l < DL3; ++l) {
            size_t o = ((size_t)(b * DL3 + l)) * C4 + c;
            m = fmaxf(m, dc32[o] * (0.5f + comp[o]));
        }
        pair[(size_t)b * (2 * C4) + c] = (_Float16)m;
    } else {
        for (int l = 0; l < PL3; ++l) {
            size_t o = ((size_t)(b * PL3 + l)) * C4 + c;
            m = fmaxf(m, pc32[o] * (0.5f + prot[o]));
        }
        pair[(size_t)b * (2 * C4) + C4 + c] = (_Float16)m;
    }
}

// ---------------- host side ----------------
static inline int gemm_blocks(int M, int N) {
    int tiles = ((M + 15) / 16) * ((N + 15) / 16);
    return (tiles + 7) / 8;
}
static inline size_t pack_elems(int K, int N) {
    return (size_t)((K + 31) / 32) * ((N + 15) / 16) * 512;
}

extern "C" void kernel_launch(void* const* d_in, const int* in_sizes, int n_in,
                              void* d_out, int out_size, void* d_ws, size_t ws_size,
                              hipStream_t stream) {
    (void)in_sizes; (void)n_in; (void)out_size; (void)ws_size;
    const int*   drug     = (const int*)  d_in[0];
    const int*   protein  = (const int*)  d_in[1];
    const float* drug_emb = (const float*)d_in[2];
    const float* prot_emb = (const float*)d_in[3];
    const float* dw1 = (const float*)d_in[4];  const float* db1 = (const float*)d_in[5];
    const float* dw2 = (const float*)d_in[6];  const float* db2 = (const float*)d_in[7];
    const float* dw3 = (const float*)d_in[8];  const float* db3 = (const float*)d_in[9];
    const float* pw1 = (const float*)d_in[10]; const float* pb1 = (const float*)d_in[11];
    const float* pw2 = (const float*)d_in[12]; const float* pb2 = (const float*)d_in[13];
    const float* pw3 = (const float*)d_in[14]; const float* pb3 = (const float*)d_in[15];
    const float* d_att_w = (const float*)d_in[16]; const float* d_att_b = (const float*)d_in[17];
    const float* p_att_w = (const float*)d_in[18]; const float* p_att_b = (const float*)d_in[19];
    const float* att_w   = (const float*)d_in[20]; const float* att_b   = (const float*)d_in[21];
    const float* fc1_w = (const float*)d_in[22]; const float* fc1_b = (const float*)d_in[23];
    const float* fc2_w = (const float*)d_in[24]; const float* fc2_b = (const float*)d_in[25];
    const float* fc3_w = (const float*)d_in[26]; const float* fc3_b = (const float*)d_in[27];
    const float* out_w = (const float*)d_in[28]; const float* out_b = (const float*)d_in[29];
    float* outp = (float*)d_out;

    // bump allocator with generous tail slack so unguarded b128 overreads stay in d_ws
    char* wp = (char*)d_ws;
    auto alloc = [&](size_t bytes) -> void* {
        void* r = (void*)wp;
        wp += ((bytes + 255) & ~(size_t)255) + 32768;
        return r;
    };
    auto h16 = [&](size_t n) { return (_Float16*)alloc(n * sizeof(_Float16)); };
    auto f32 = [&](size_t n) { return (float*)alloc(n * sizeof(float)); };

    _Float16* de16  = h16((size_t)BB * LD * DIMD);
    _Float16* pe16  = h16((size_t)BB * LP * DIMD);
    _Float16* d1_16 = h16((size_t)BB * DL1 * CONVC);
    _Float16* d2_16 = h16((size_t)BB * DL2 * 2 * CONVC);
    _Float16* dc16  = h16((size_t)BB * DL3 * C4);
    float*    dc32  = f32((size_t)BB * DL3 * C4);
    _Float16* p1_16 = h16((size_t)BB * PL1 * CONVC);
    _Float16* p2_16 = h16((size_t)BB * PL2 * 2 * CONVC);
    _Float16* pc16  = h16((size_t)BB * PL3 * C4);
    float*    pc32  = f32((size_t)BB * PL3 * C4);
    _Float16* dw1P = h16(pack_elems(4 * DIMD, CONVC));
    _Float16* dw2P = h16(pack_elems(6 * CONVC, 2 * CONVC));
    _Float16* dw3P = h16(pack_elems(8 * 2 * CONVC, C4));
    _Float16* pw1P = h16(pack_elems(4 * DIMD, CONVC));
    _Float16* pw2P = h16(pack_elems(8 * CONVC, 2 * CONVC));
    _Float16* pw3P = h16(pack_elems(12 * 2 * CONVC, C4));
    _Float16* dawP = h16(pack_elems(C4, C4));
    _Float16* pawP = h16(pack_elems(C4, C4));
    _Float16* awP  = h16(pack_elems(C4, C4));
    _Float16* fc1P = h16(pack_elems(2 * C4, 1024));
    _Float16* fc2P = h16(pack_elems(1024, 1024));
    _Float16* fc3P = h16(pack_elems(1024, 512));
    _Float16* owP  = h16(pack_elems(512, 2));
    float* datt32 = f32((size_t)BB * DL3 * C4);
    float* patt32 = f32((size_t)BB * PL3 * C4);
    _Float16* rowm16 = h16((size_t)BB * DL3 * C4);
    _Float16* colm16 = h16((size_t)BB * PL3 * C4);
    float* comp32 = f32((size_t)BB * DL3 * C4);
    float* prot32 = f32((size_t)BB * PL3 * C4);
    _Float16* pair16 = h16((size_t)BB * 2 * C4);
    _Float16* h1 = h16((size_t)BB * 1024);
    _Float16* h2 = h16((size_t)BB * 1024);
    _Float16* h3 = h16((size_t)BB * 512);
    (void)alloc(65536);   // final slack for epilogue-row overreads

    const int T = 256;
    auto nb = [](size_t n) { return (int)((n + 255) / 256); };

    // ---- pack weights into WMMA fragment-major B layout ----
    auto packc = [&](const float* w, _Float16* Bp, int Cout, int Cin, int Kw) {
        int KT = (Kw * Cin + 31) / 32, NT = (Cout + 15) / 16;
        pack_w_conv<<<nb((size_t)KT * NT * 512), T, 0, stream>>>(w, Bp, Cout, Cin, Kw, KT, NT);
    };
    auto packd = [&](const float* w, _Float16* Bp, int K, int N) {
        int KT = (K + 31) / 32, NT = (N + 15) / 16;
        pack_w_dense<<<nb((size_t)KT * NT * 512), T, 0, stream>>>(w, Bp, K, N, KT, NT);
    };
    packc(dw1, dw1P, CONVC, DIMD, 4);
    packc(dw2, dw2P, 2 * CONVC, CONVC, 6);
    packc(dw3, dw3P, C4, 2 * CONVC, 8);
    packc(pw1, pw1P, CONVC, DIMD, 4);
    packc(pw2, pw2P, 2 * CONVC, CONVC, 8);
    packc(pw3, pw3P, C4, 2 * CONVC, 12);
    packd(d_att_w, dawP, C4, C4);
    packd(p_att_w, pawP, C4, C4);
    packd(att_w,   awP,  C4, C4);
    packd(fc1_w, fc1P, 2 * C4, 1024);
    packd(fc2_w, fc2P, 1024, 1024);
    packd(fc3_w, fc3P, 1024, 512);
    packd(out_w, owP, 512, 2);

    // ---- embeddings (channel-last) ----
    embed_gather<<<nb((size_t)BB * LD * DIMD), T, 0, stream>>>(drug, drug_emb, de16, BB * LD, DIMD);
    embed_gather<<<nb((size_t)BB * LP * DIMD), T, 0, stream>>>(protein, prot_emb, pe16, BB * LP, DIMD);

    // ---- conv stacks (implicit-GEMM WMMA, fused ReLU) ----
    conv_gemm_wmma<4 * DIMD><<<gemm_blocks(BB * DL1, CONVC), T, 0, stream>>>(
        de16, dw1P, db1, d1_16, (float*)nullptr, BB, LD, DL1, DIMD, CONVC);
    conv_gemm_wmma<6 * CONVC><<<gemm_blocks(BB * DL2, 2 * CONVC), T, 0, stream>>>(
        d1_16, dw2P, db2, d2_16, (float*)nullptr, BB, DL1, DL2, CONVC, 2 * CONVC);
    conv_gemm_wmma<8 * 2 * CONVC><<<gemm_blocks(BB * DL3, C4), T, 0, stream>>>(
        d2_16, dw3P, db3, dc16, dc32, BB, DL2, DL3, 2 * CONVC, C4);
    conv_gemm_wmma<4 * DIMD><<<gemm_blocks(BB * PL1, CONVC), T, 0, stream>>>(
        pe16, pw1P, pb1, p1_16, (float*)nullptr, BB, LP, PL1, DIMD, CONVC);
    conv_gemm_wmma<8 * CONVC><<<gemm_blocks(BB * PL2, 2 * CONVC), T, 0, stream>>>(
        p1_16, pw2P, pb2, p2_16, (float*)nullptr, BB, PL1, PL2, CONVC, 2 * CONVC);
    conv_gemm_wmma<12 * 2 * CONVC><<<gemm_blocks(BB * PL3, C4), T, 0, stream>>>(
        p2_16, pw3P, pb3, pc16, pc32, BB, PL2, PL3, 2 * CONVC, C4);

    // ---- attention linears (WMMA) ----
    gemm_f16_wmma<C4><<<gemm_blocks(BB * DL3, C4), T, 0, stream>>>(
        dc16, dawP, d_att_b, datt32, (_Float16*)nullptr, BB * DL3, C4, ACT_NONE);
    gemm_f16_wmma<C4><<<gemm_blocks(BB * PL3, C4), T, 0, stream>>>(
        pc16, pawP, p_att_b, patt32, (_Float16*)nullptr, BB * PL3, C4, ACT_NONE);

    // ---- fused means over R ----
    attn_rowmean<<<BB * DL3, C4, 0, stream>>>(datt32, patt32, rowm16);
    attn_colmean<<<BB * PL3, C4, 0, stream>>>(datt32, patt32, colm16);

    // ---- mean @ att_w + b, sigmoid (WMMA) ----
    gemm_f16_wmma<C4><<<gemm_blocks(BB * DL3, C4), T, 0, stream>>>(
        rowm16, awP, att_b, comp32, (_Float16*)nullptr, BB * DL3, C4, ACT_SIGM);
    gemm_f16_wmma<C4><<<gemm_blocks(BB * PL3, C4), T, 0, stream>>>(
        colm16, awP, att_b, prot32, (_Float16*)nullptr, BB * PL3, C4, ACT_SIGM);

    // ---- gate + global max pool ----
    gate_pool<<<2 * BB, C4, 0, stream>>>(dc32, comp32, pc32, prot32, pair16);

    // ---- FC head (WMMA, leaky-relu 0.01) ----
    gemm_f16_wmma<2 * C4><<<gemm_blocks(BB, 1024), T, 0, stream>>>(
        pair16, fc1P, fc1_b, (float*)nullptr, h1, BB, 1024, ACT_LREL);
    gemm_f16_wmma<1024><<<gemm_blocks(BB, 1024), T, 0, stream>>>(
        h1, fc2P, fc2_b, (float*)nullptr, h2, BB, 1024, ACT_LREL);
    gemm_f16_wmma<1024><<<gemm_blocks(BB, 512), T, 0, stream>>>(
        h2, fc3P, fc3_b, (float*)nullptr, h3, BB, 512, ACT_LREL);
    gemm_f16_wmma<512><<<gemm_blocks(BB, 2), T, 0, stream>>>(
        h3, owP, out_b, outp, (_Float16*)nullptr, BB, 2, ACT_NONE);
}
